// AttentionBlock_1451698946268
// MI455X (gfx1250) — matrix operand order repaired and checked by reading, not compiled, not run on hardware
//
#include <hip/hip_runtime.h>
#include <hip/hip_bf16.h>

// ---------------- problem constants ----------------
#define DDIM 1024
#define NHEAD 16
#define HDIM 64
#define BB 2
#define TT 2048
#define BT (BB * TT)          // 4096 tokens
#define D3 (3 * DDIM)         // 3072

typedef __attribute__((ext_vector_type(16))) __bf16 bf16x16;
typedef __attribute__((ext_vector_type(8)))  __bf16 bf16x8;
typedef __attribute__((ext_vector_type(8)))  float  f32x8;

union bfrag { bf16x16 v; bf16x8 h[2]; };

__device__ __forceinline__ unsigned short f2bf(float f) {
  unsigned int u = __float_as_uint(f);
  u += 0x7FFFu + ((u >> 16) & 1u);      // round-to-nearest-even
  return (unsigned short)(u >> 16);
}

__device__ __forceinline__ f32x8 wmma_bf16(bf16x16 a, bf16x16 b, f32x8 c) {
  return __builtin_amdgcn_wmma_f32_16x16x32_bf16(false, a, false, b, (short)0, c,
                                                 false, false);
}

// A-fragment (16x32): per-lane 8-elem chunks at k = hi*8 and 16+hi*8
__device__ __forceinline__ bfrag loadA_frag(const unsigned short* p, int kk) {
  bfrag f;
  f.h[0] = *reinterpret_cast<const bf16x8*>(p + kk);
  f.h[1] = *reinterpret_cast<const bf16x8*>(p + kk + 16);
  return f;
}
// B-fragment (32x16): per-lane 8-elem chunks at k = hi*8 and 16+hi*8
__device__ __forceinline__ bfrag loadB_frag(const unsigned short* p, int kk) {
  bfrag f;
  f.h[0] = *reinterpret_cast<const bf16x8*>(p + kk);
  f.h[1] = *reinterpret_cast<const bf16x8*>(p + kk + 16);
  return f;
}

// gfx1250 async copy: global -> LDS, 16 bytes per lane, tracked by ASYNCcnt
__device__ __forceinline__ void async_load_b128(unsigned lds_off,
                                                const void* gaddr) {
  asm volatile("global_load_async_to_lds_b128 %0, %1, off"
               :: "v"(lds_off), "v"(gaddr) : "memory");
}
__device__ __forceinline__ void wait_asynccnt0() {
  asm volatile("s_wait_asynccnt 0" ::: "memory");
}

// ---------------- prep kernels ----------------
__global__ void k_f32_to_bf16(const float* __restrict__ in,
                              unsigned short* __restrict__ out, int n) {
  int i = blockIdx.x * blockDim.x + threadIdx.x;
  if (i < n) out[i] = f2bf(in[i]);
}

// in: [K, Nn] row-major f32   ->  out: [Nn, K] bf16 (transposed)
__global__ void k_transpose_bf16(const float* __restrict__ in,
                                 unsigned short* __restrict__ out,
                                 int K, int Nn) {
  int i = blockIdx.x * blockDim.x + threadIdx.x;
  if (i >= K * Nn) return;
  int n = i % Nn;
  int k = i / Nn;
  out[(size_t)n * K + k] = f2bf(in[i]);
}

// ---------------- shared WMMA mainloop: 32x64 tile, double-buffered ---------
__device__ __forceinline__ void gemm_mainloop_32x64(
    const unsigned short* __restrict__ A, const unsigned short* __restrict__ Bt,
    int K, int mBase, int nBase, int m16, int hi,
    f32x8 (&accA)[4], f32x8 (&accB)[4]) {
  const unsigned short* aP0 = A + (size_t)(mBase + m16) * K + hi * 8;
  const unsigned short* aP1 = A + (size_t)(mBase + 16 + m16) * K + hi * 8;
  const unsigned short* bP[4];
#pragma unroll
  for (int j = 0; j < 4; ++j)
    bP[j] = Bt + (size_t)(nBase + j * 16 + m16) * K + hi * 8;

  bfrag a0c = loadA_frag(aP0, 0);
  bfrag a1c = loadA_frag(aP1, 0);
  bfrag bc[4];
#pragma unroll
  for (int j = 0; j < 4; ++j) bc[j] = loadB_frag(bP[j], 0);

  for (int kk = 0; kk < K; kk += 32) {
    // near-cache speculative prefetch 16 steps (512 elements) ahead (fault-safe)
    __builtin_prefetch(aP0 + kk + 512, 0, 3);
    __builtin_prefetch(bP[0] + kk + 512, 0, 3);

    const int kn = (kk + 32 < K) ? kk + 32 : 0;   // next-step (wraps harmlessly)
    bfrag a0n = loadA_frag(aP0, kn);
    bfrag a1n = loadA_frag(aP1, kn);
    bfrag bn[4];
#pragma unroll
    for (int j = 0; j < 4; ++j) bn[j] = loadB_frag(bP[j], kn);

#pragma unroll
    for (int j = 0; j < 4; ++j) accA[j] = wmma_bf16(a0c.v, bc[j].v, accA[j]);
#pragma unroll
    for (int j = 0; j < 4; ++j) accB[j] = wmma_bf16(a1c.v, bc[j].v, accB[j]);

    a0c = a0n; a1c = a1n;
#pragma unroll
    for (int j = 0; j < 4; ++j) bc[j] = bn[j];
  }
}

// ---------------- fused QKV GEMM: xbf @ w_attn^T + b -> packed bf16 Q/K/V^T
__global__ __launch_bounds__(32)
void k_gemm_qkv(const unsigned short* __restrict__ A,   // xbf [BT, D]
                const unsigned short* __restrict__ Bt,  // wAT [3D, D]
                const float* __restrict__ bias,         // [3D]
                unsigned short* __restrict__ Qb,        // [B,NH,T,H] (pre-scaled)
                unsigned short* __restrict__ Kb,        // [B,NH,T,H]
                unsigned short* __restrict__ Vt) {      // [B,NH,H,T]
  const int lane = threadIdx.x & 31;
  const int m16 = lane & 15, hi = lane >> 4;
  const int mBase = blockIdx.x * 32;
  const int nBase = blockIdx.y * 64;

  f32x8 accA[4] = {}, accB[4] = {};
  gemm_mainloop_32x64(A, Bt, DDIM, mBase, nBase, m16, hi, accA, accB);

  // 64-wide column tile = exactly one section (q/k/v) and one head (uniform)
  const int sect = nBase >> 10;           // 0=q, 1=k, 2=v
  const int head = (nBase & 1023) >> 6;

#pragma unroll
  for (int j = 0; j < 4; ++j) {
    const int hd = j * 16 + m16;
    const float bv = bias[nBase + j * 16 + m16];
#pragma unroll
    for (int set = 0; set < 2; ++set) {
#pragma unroll
      for (int r = 0; r < 8; ++r) {
        const int row = mBase + set * 16 + r + 8 * hi;   // token index
        const float val = (set ? accB[j][r] : accA[j][r]) + bv;
        const int b = row >> 11;
        const int t = row & 2047;
        const size_t bh = (size_t)(b * NHEAD + head);
        if (sect == 0)
          Qb[(bh * TT + t) * HDIM + hd] = f2bf(val * 0.125f);  // 1/sqrt(64)
        else if (sect == 1)
          Kb[(bh * TT + t) * HDIM + hd] = f2bf(val);
        else
          Vt[(bh * HDIM + hd) * TT + t] = f2bf(val);
      }
    }
  }
}

// ---------------- projection GEMM: ob @ w_proj^T + b -> f32 out -------------
__global__ __launch_bounds__(32)
void k_gemm_bias(const unsigned short* __restrict__ A,
                 const unsigned short* __restrict__ Bt,
                 const float* __restrict__ bias,
                 float* __restrict__ C, int M, int Nn, int K) {
  const int lane = threadIdx.x & 31;
  const int m16 = lane & 15, hi = lane >> 4;
  const int mBase = blockIdx.x * 32;
  const int nBase = blockIdx.y * 64;

  f32x8 accA[4] = {}, accB[4] = {};
  gemm_mainloop_32x64(A, Bt, K, mBase, nBase, m16, hi, accA, accB);

#pragma unroll
  for (int j = 0; j < 4; ++j) {
    const int col = nBase + j * 16 + m16;
    const float bv = bias[col];
#pragma unroll
    for (int r = 0; r < 8; ++r) {
      C[(size_t)(mBase + r + 8 * hi) * Nn + col] = accA[j][r] + bv;
      C[(size_t)(mBase + 16 + r + 8 * hi) * Nn + col] = accB[j][r] + bv;
    }
  }
}

// ---------------- cooperative flash attention ------------------------------
// Block = 4 waves x 32 lanes; block owns 128 consecutive queries of one (b,h).
// Each wave handles 32 queries (two 16-row tiles). K/V tiles (32 keys) are
// async-DMA'd into double-buffered LDS once per block and shared by all waves.
#define ATT_WAVES 4
#define KPAD 72    // 64 bf16 + pad (row stride, bf16 units)
#define VPAD 40    // 32 bf16 + pad

__global__ __launch_bounds__(32 * ATT_WAVES)
void k_attention(const unsigned short* __restrict__ Qb,
                 const unsigned short* __restrict__ Kb,
                 const unsigned short* __restrict__ Vt,
                 unsigned short* __restrict__ Ob /* [BT, D] bf16 */) {
  __shared__ unsigned short shK[2][32][KPAD];        // keys x hdim
  __shared__ unsigned short shV[2][HDIM][VPAD];      // hdim x keys
  __shared__ unsigned short shP[ATT_WAVES][32][VPAD];

  const int tid  = threadIdx.x;
  const int wave = tid >> 5;
  const int lane = tid & 31;
  const int m16 = lane & 15, hi = lane >> 4;

  const int qblk = blockIdx.x & 15;        // 128-query block within sequence
  const int bh   = blockIdx.x >> 4;
  const int b    = bh >> 4;
  const int h    = bh & 15;
  const int blockQ  = qblk * 128;
  const int qLoBase = blockQ + wave * 32;  // this wave's first query tile

  const unsigned short* Qp = Qb + ((size_t)bh * TT + qLoBase) * HDIM;
  const unsigned short* Kp = Kb + (size_t)bh * TT * HDIM;
  const unsigned short* Vp = Vt + (size_t)bh * HDIM * TT;

  // Q: two 16-row tiles, each two hdim-chunk A-fragments
  bfrag aq[2][2];
#pragma unroll
  for (int t = 0; t < 2; ++t) {
    const unsigned short* qrow = Qp + (t * 16 + m16) * HDIM + hi * 8;
    aq[t][0] = loadA_frag(qrow, 0);
    aq[t][1] = loadA_frag(qrow, 32);
  }

  f32x8 O[2][4] = {};
  float mrow[2][8], lsum[2][8];
#pragma unroll
  for (int t = 0; t < 2; ++t)
#pragma unroll
    for (int r = 0; r < 8; ++r) { mrow[t][r] = -3.0e38f; lsum[t][r] = 0.0f; }

  const unsigned ldsKb[2] = {(unsigned)(size_t)&shK[0][0][0],
                             (unsigned)(size_t)&shK[1][0][0]};
  const unsigned ldsVb[2] = {(unsigned)(size_t)&shV[0][0][0],
                             (unsigned)(size_t)&shV[1][0][0]};

  // async-stage one 32-key K/V tile into LDS buffer `buf`
  auto stage_kv = [&](int buf, int s0) {
    // K tile: 32 rows x 128B = 256 x 16B chunks, 2 per thread
#pragma unroll
    for (int u = 0; u < 2; ++u) {
      int c = tid * 2 + u;
      int row = c >> 3, bo = (c & 7) * 16;
      const char* g = (const char*)Kp + (size_t)(s0 + row) * 128 + bo;
      async_load_b128(ldsKb[buf] + row * (KPAD * 2) + bo, g);
    }
    // V tile: 64 rows x 64B = 256 x 16B chunks, 2 per thread
#pragma unroll
    for (int u = 0; u < 2; ++u) {
      int c = tid * 2 + u;
      int row = c >> 2, bo = (c & 3) * 16;
      const char* g = (const char*)Vp + (size_t)row * (TT * 2) +
                      (size_t)s0 * 2 + bo;
      async_load_b128(ldsVb[buf] + row * (VPAD * 2) + bo, g);
    }
  };

  const int nb = 4 * (qblk + 1);           // 32-key blocks this workgroup needs
  stage_kv(0, 0);

  for (int i = 0; i < nb; ++i) {
    const int s0 = i * 32;
    const int buf = i & 1;
    wait_asynccnt0();                      // our async stores to LDS done
    __syncthreads();                       // everyone's staging visible
    if (i + 1 < nb) stage_kv(buf ^ 1, s0 + 32);   // overlap DMA with compute

    if (s0 <= qLoBase) {                   // wave-uniform -> EXEC stays all-1
      // K B-fragments from LDS (shared by both query tiles)
      bfrag kb[2][2];
#pragma unroll
      for (int g = 0; g < 2; ++g)
#pragma unroll
        for (int c = 0; c < 2; ++c)
          kb[g][c] = loadB_frag(&shK[buf][g * 16 + m16][0] + c * 32 + hi * 8, 0);

#pragma unroll
      for (int t = 0; t < 2; ++t) {
        f32x8 S0 = {}, S1 = {};
        S0 = wmma_bf16(aq[t][0].v, kb[0][0].v, S0);
        S0 = wmma_bf16(aq[t][1].v, kb[0][1].v, S0);
        S1 = wmma_bf16(aq[t][0].v, kb[1][0].v, S1);
        S1 = wmma_bf16(aq[t][1].v, kb[1][1].v, S1);

        const int qBase = qLoBase + t * 16;
#pragma unroll
        for (int r = 0; r < 8; ++r) {
          int tq = qBase + r + 8 * hi;
          float v0 = ((s0 + m16) <= tq)      ? S0[r] : -3.0e38f;
          float v1 = ((s0 + 16 + m16) <= tq) ? S1[r] : -3.0e38f;
          float mx = fmaxf(v0, v1);
          mx = fmaxf(mx, __shfl_xor(mx, 1, 16));
          mx = fmaxf(mx, __shfl_xor(mx, 2, 16));
          mx = fmaxf(mx, __shfl_xor(mx, 4, 16));
          mx = fmaxf(mx, __shfl_xor(mx, 8, 16));
          float mnew  = fmaxf(mrow[t][r], mx);
          float scale = __expf(mrow[t][r] - mnew);
          mrow[t][r] = mnew;
          float p0 = __expf(v0 - mnew);
          float p1 = __expf(v1 - mnew);
          lsum[t][r] = lsum[t][r] * scale + p0 + p1;
          O[t][0][r] *= scale; O[t][1][r] *= scale;
          O[t][2][r] *= scale; O[t][3][r] *= scale;
          shP[wave][t * 16 + r + 8 * hi][m16]      = f2bf(p0);
          shP[wave][t * 16 + r + 8 * hi][m16 + 16] = f2bf(p1);
        }
      }
      asm volatile("s_wait_dscnt 0" ::: "memory");

      // V B-fragments from LDS (shared by both tiles)
      bfrag bv[4];
#pragma unroll
      for (int j = 0; j < 4; ++j)
        bv[j] = loadB_frag(&shV[buf][j * 16 + m16][0] + hi * 8, 0);

#pragma unroll
      for (int t = 0; t < 2; ++t) {
        bfrag ap = loadA_frag(&shP[wave][t * 16 + m16][0] + hi * 8, 0);
#pragma unroll
        for (int j = 0; j < 4; ++j) O[t][j] = wmma_bf16(ap.v, bv[j].v, O[t][j]);
      }
    }
  }

  // ---- epilogue: normalize rows, emit bf16 [BT, D] ----
#pragma unroll
  for (int t = 0; t < 2; ++t) {
#pragma unroll
    for (int r = 0; r < 8; ++r) {
      float s = lsum[t][r];
      s += __shfl_xor(s, 1, 16);
      s += __shfl_xor(s, 2, 16);
      s += __shfl_xor(s, 4, 16);
      s += __shfl_xor(s, 8, 16);
      float rinv = 1.0f / s;
      int tq = qLoBase + t * 16 + r + 8 * hi;
#pragma unroll
      for (int j = 0; j < 4; ++j) {
        size_t idx = (size_t)(b * TT + tq) * DDIM + h * HDIM + j * 16 + m16;
        Ob[idx] = f2bf(O[t][j][r] * rinv);
      }
    }
  }
}

// ---------------- host side ----------------
extern "C" void kernel_launch(void* const* d_in, const int* in_sizes, int n_in,
                              void* d_out, int out_size, void* d_ws,
                              size_t ws_size, hipStream_t stream) {
  (void)in_sizes; (void)n_in; (void)out_size; (void)ws_size;
  const float* x      = (const float*)d_in[0];
  const float* w_attn = (const float*)d_in[1];
  const float* b_attn = (const float*)d_in[2];
  const float* w_proj = (const float*)d_in[3];
  const float* b_proj = (const float*)d_in[4];
  float* out = (float*)d_out;

  char* ws = (char*)d_ws;
  size_t off = 0;
  auto take = [&](size_t bytes) {
    char* p = ws + off;
    off = (off + bytes + 255) & ~(size_t)255;
    return p;
  };
  unsigned short* xbf = (unsigned short*)take((size_t)BT * DDIM * 2);
  unsigned short* wAT = (unsigned short*)take((size_t)D3 * DDIM * 2);
  unsigned short* wPT = (unsigned short*)take((size_t)DDIM * DDIM * 2);
  unsigned short* qb  = (unsigned short*)take((size_t)BT * DDIM * 2);
  unsigned short* kb  = (unsigned short*)take((size_t)BT * DDIM * 2);
  unsigned short* vt  = (unsigned short*)take((size_t)BT * DDIM * 2);
  unsigned short* ob  = (unsigned short*)take((size_t)BT * DDIM * 2);

  // 1) fp32 -> bf16 conversions / weight transposes
  k_f32_to_bf16<<<(BT * DDIM) / 256, 256, 0, stream>>>(x, xbf, BT * DDIM);
  k_transpose_bf16<<<(DDIM * D3) / 256, 256, 0, stream>>>(w_attn, wAT, DDIM, D3);
  k_transpose_bf16<<<(DDIM * DDIM) / 256, 256, 0, stream>>>(w_proj, wPT, DDIM, DDIM);

  // 2) fused QKV GEMM -> packed bf16 Q (scaled), K, V^T
  k_gemm_qkv<<<dim3(BT / 32, D3 / 64), 32, 0, stream>>>(xbf, wAT, b_attn,
                                                        qb, kb, vt);

  // 3) cooperative causal flash attention (async K/V staging, bf16 out)
  k_attention<<<BB * NHEAD * (TT / 128), 32 * ATT_WAVES, 0, stream>>>(qb, kb,
                                                                      vt, ob);

  // 4) output projection GEMM -> d_out
  k_gemm_bias<<<dim3(BT / 32, DDIM / 64), 32, 0, stream>>>(ob, wPT, b_proj, out,
                                                           BT, DDIM, DDIM);
}
